// Block_7584912245089
// MI455X (gfx1250) — matrix operand order repaired and checked by reading, not compiled, not run on hardware
//
#include <hip/hip_runtime.h>
#include <hip/hip_bf16.h>
#include <math.h>

// ---------------- CDNA5 WMMA types ----------------
typedef __attribute__((ext_vector_type(16))) __bf16 v16bf;
typedef __attribute__((ext_vector_type(8)))  float  v8f;

// ---------------- problem constants ----------------
#define BB     2
#define HHH    64
#define WWW    128
#define HWPIX  (HHH*WWW)            // 8192
#define NPB    (HWPIX+2)            // 8194 nodes per batch (2 poles first)
#define NNODE  (BB*NPB)             // 16388
#define NPADR  16400                // NNODE padded up to full 16-row tiles
#define EE     131072
#define ENTOT  (EE + NNODE)         // 147460 (edges + self loops)
#define HCC    512                  // heads*hid
#define PIX    (BB*HWPIX)           // 16384
#define CXN    117
#define TGTN   39
#define LRELU  0.01f
#define ASLOPE 0.2f

// ---------------- small device helpers ----------------
__device__ inline unsigned short f2bf(float f) {        // RNE f32 -> bf16
    unsigned int u = __float_as_uint(f);
    u = (u + 0x7FFFu + ((u >> 16) & 1u)) >> 16;
    return (unsigned short)u;
}

__device__ inline float lrelu(float v, float s) { return v >= 0.f ? v : s * v; }

__device__ inline void atomicMaxF(float* a, float v) {
    if (v >= 0.f) atomicMax((int*)a, __float_as_int(v));
    else          atomicMin((unsigned int*)a, (unsigned int)__float_as_int(v));
}

struct Q2 { uint4 a, b; };   // 32B, bit-cast into v16bf (SROA-proof, no stack)

// A fragment, ISA 16-bit A 16x32 layout: lane half h=lane>>4 holds
// K = [h*8, h*8+8) and [16+h*8, 16+h*8+8)  (two contiguous 16B chunks)
// UNCONDITIONAL loads: A buffers are padded+zeroed, always in-bounds.
__device__ inline v16bf ldA(const unsigned short* __restrict__ row, int kk, int h) {
    Q2 q;
    q.a = *(const uint4*)(row + kk + h * 8);
    q.b = *(const uint4*)(row + kk + 16 + h * 8);
    return __builtin_bit_cast(v16bf, q);
}

// B fragment from B^T (N x K row-major): lane n=lane&15, K = [h*8, h*8+8) and [16+h*8, 16+h*8+8)
__device__ inline v16bf ldB(const unsigned short* __restrict__ rowT, int kk, int h) {
    Q2 q;
    q.a = *(const uint4*)(rowT + kk + h * 8);
    q.b = *(const uint4*)(rowT + kk + 16 + h * 8);
    return __builtin_bit_cast(v16bf, q);
}

// ---------------- generic WMMA GEMM ----------------
// C[M,N] = act( A[Mpad,K]bf16 @ (Bt[Npad,K]bf16)^T + bias ), optional f32/bf16 out.
// A must be valid (zero-padded) through the last 16-row tile covering M.
__global__ void __launch_bounds__(128)
k_gemm_wmma(const unsigned short* __restrict__ A, int lda,
            const unsigned short* __restrict__ Bt, int ldb,
            const float* __restrict__ bias,
            float* __restrict__ Cf, int ldc,
            unsigned short* __restrict__ Cb, int ldcb,
            int M, int N, int K, int act)
{
    const int lane = threadIdx.x & 31;
    const int wave = threadIdx.x >> 5;
    const int tileM = blockIdx.x * 4 + wave;
    const int row0 = tileM * 16;
    const int col0 = blockIdx.y * 16;
    if (row0 >= M) return;                       // wave-uniform exit

    const int h = lane >> 4;                     // half-wave id
    const unsigned short* arow = A + (size_t)(row0 + (lane & 15)) * lda;
    const unsigned short* brow = Bt + (size_t)(col0 + (lane & 15)) * ldb;

    v8f acc = {0.f, 0.f, 0.f, 0.f, 0.f, 0.f, 0.f, 0.f};
    for (int kk = 0; kk < K; kk += 32) {
        __builtin_prefetch(arow + kk + 128, 0, 3);          // global_prefetch_b8
        v16bf af = ldA(arow, kk, h);
        v16bf bf = ldB(brow, kk, h);
        acc = __builtin_amdgcn_wmma_f32_16x16x32_bf16(
                  false, af, false, bf, (short)0, acc, false, false);
    }

    const int colg = col0 + (lane & 15);
    const float bv = (bias && colg < N) ? bias[colg] : 0.f;
#pragma unroll
    for (int r = 0; r < 8; ++r) {
        int rowg = row0 + r + h * 8;
        if (rowg < M && colg < N) {
            float v = acc[r] + bv;
            if (act) v = lrelu(v, LRELU);
            if (Cf) Cf[(size_t)rowg * ldc + colg] = v;
            if (Cb) Cb[(size_t)rowg * ldcb + colg] = f2bf(v);
        }
    }
}

// ---------------- conv as implicit GEMM (9 shifted taps) ----------------
// lm: [B,66,130,512] bf16 channels-last; wct: [9,128,512] bf16 (tap, oc, ic)
// writes leaky(BN(conv+bias)) as bf16 into zin[:, 512:640]
__global__ void __launch_bounds__(128)
k_conv_wmma(const unsigned short* __restrict__ lm,
            const unsigned short* __restrict__ wct,
            const float* __restrict__ conv_b,
            const float* __restrict__ bn_g, const float* __restrict__ bn_b,
            unsigned short* __restrict__ zin)
{
    const int lane = threadIdx.x & 31;
    const int wave = threadIdx.x >> 5;
    const int tileM = blockIdx.x * 4 + wave;     // 16 pixels per tile
    const int row0 = tileM * 16;
    const int col0 = blockIdx.y * 16;            // output channel tile
    const int h = lane >> 4;

    const int m  = row0 + (lane & 15);           // pixel id (always < PIX)
    const int b  = m >> 13;
    const int p  = m & (HWPIX - 1);
    const int oy = p >> 7;
    const int ox = p & (WWW - 1);
    const int oc = col0 + (lane & 15);

    v8f acc = {0.f, 0.f, 0.f, 0.f, 0.f, 0.f, 0.f, 0.f};
    for (int tap = 0; tap < 9; ++tap) {
        const int ky = tap / 3, kx = tap % 3;
        const unsigned short* arow =
            lm + ((size_t)((b * 66 + oy + ky) * 130 + (ox + kx))) * HCC;
        const unsigned short* brow = wct + ((size_t)tap * 128 + oc) * HCC;
        for (int kk = 0; kk < HCC; kk += 32) {
            v16bf af = ldA(arow, kk, h);
            v16bf bf = ldB(brow, kk, h);
            acc = __builtin_amdgcn_wmma_f32_16x16x32_bf16(
                      false, af, false, bf, (short)0, acc, false, false);
        }
    }

    const float scale = bn_g[oc] * rsqrtf(1.f + 1e-5f);
    const float shift = bn_b[oc];
    const float cb = conv_b[oc];
#pragma unroll
    for (int r = 0; r < 8; ++r) {
        int rowg = row0 + r + h * 8;
        float v = lrelu((acc[r] + cb) * scale + shift, LRELU);
        zin[(size_t)rowg * 640 + 512 + oc] = f2bf(v);
    }
}

// ---------------- staging / elementwise kernels ----------------
__global__ void k_fill_u32(unsigned int* p, size_t n, unsigned int v) {
    size_t i = (size_t)blockIdx.x * blockDim.x + threadIdx.x;
    for (; i < n; i += (size_t)gridDim.x * blockDim.x) p[i] = v;
}

// node features: [NPADR,128] bf16 (poles + pad rows zero, 123 real feats)
__global__ void k_build_nodes(const float* __restrict__ x, const float* __restrict__ xc,
                              unsigned short* __restrict__ nodes) {
    int idx = blockIdx.x * blockDim.x + threadIdx.x;
    if (idx >= NPADR * 128) return;
    int c = idx & 127, n = idx >> 7;
    float v = 0.f;
    if (n < NNODE) {
        int b = n / NPB, i = n % NPB;
        if (i >= 2 && c < 123) {
            int p = i - 2, y = p >> 7, xcoord = p & 127;
            if (c < CXN) v = x[(((size_t)b * CXN + c) * HHH + y) * WWW + xcoord];
            else         v = xc[(((size_t)b * 6 + (c - CXN)) * HHH + y) * WWW + xcoord];
        }
    }
    nodes[idx] = f2bf(v);
}

// W[K,N] f32 -> Bt[NP,KP] bf16 (transposed + zero-padded)
__global__ void k_wtrans(const float* __restrict__ Wk, int K, int N,
                         unsigned short* __restrict__ Bt, int KP, int NP) {
    int idx = blockIdx.x * blockDim.x + threadIdx.x;
    if (idx >= NP * KP) return;
    int kp = idx % KP, np = idx / KP;
    float v = (np < N && kp < K) ? Wk[(size_t)kp * N + np] : 0.f;
    Bt[idx] = f2bf(v);
}

// conv_w OIHW [128,512,3,3] -> wct [9,128,512] bf16
__global__ void k_convtrans(const float* __restrict__ cw, unsigned short* __restrict__ wct) {
    int idx = blockIdx.x * blockDim.x + threadIdx.x;
    if (idx >= 9 * 128 * 512) return;
    int ic = idx & 511, r = idx >> 9;
    int oc = r & 127, tap = r >> 7;
    int ky = tap / 3, kx = tap % 3;
    wct[idx] = f2bf(cw[(((size_t)oc * 512 + ic) * 3 + ky) * 3 + kx]);
}

// w_eh[j][h] = sum_c We[j, h*128+c] * ae[h,c]   (32 threads)
__global__ void k_weh(const float* __restrict__ We, const float* __restrict__ ae,
                      float* __restrict__ weh) {
    int t = threadIdx.x;
    if (t >= 32) return;
    int hh = t & 3, j = t >> 2;
    float s = 0.f;
    for (int c = 0; c < 128; ++c) s += We[(size_t)j * 512 + hh * 128 + c] * ae[hh * 128 + c];
    weh[j * 4 + hh] = s;
}

// a_src/a_dst dots per (n,h)
__global__ void k_attdot(const float* __restrict__ xp, const float* __restrict__ as,
                         const float* __restrict__ ad, float* __restrict__ a_src,
                         float* __restrict__ a_dst) {
    int idx = blockIdx.x * blockDim.x + threadIdx.x;
    if (idx >= NNODE * 4) return;
    int hh = idx & 3, n = idx >> 2;
    const float* row = xp + (size_t)n * HCC + hh * 128;
    float ss = 0.f, sd = 0.f;
    for (int c = 0; c < 128; ++c) { float v = row[c]; ss += v * as[hh * 128 + c]; sd += v * ad[hh * 128 + c]; }
    a_src[idx] = ss; a_dst[idx] = sd;
}

// self-loop attr accumulation
__global__ void k_loopsum(const int* __restrict__ dst, const float* __restrict__ ea,
                          float* __restrict__ cnt, float* __restrict__ lsum) {
    int e = blockIdx.x * blockDim.x + threadIdx.x;
    if (e >= EE) return;
    int d = dst[e];
    atomicAdd(&cnt[d], 1.f);
    for (int j = 0; j < 8; ++j) atomicAdd(&lsum[(size_t)d * 8 + j], ea[(size_t)e * 8 + j]);
}
__global__ void k_loopdiv(float* __restrict__ lsum, const float* __restrict__ cnt) {
    int idx = blockIdx.x * blockDim.x + threadIdx.x;
    if (idx >= NNODE * 8) return;
    lsum[idx] /= fmaxf(cnt[idx >> 3], 1.f);
}

__device__ inline void edge_sd(int e, const int* src, const int* dst, int& s, int& d) {
    if (e < EE) { s = src[e]; d = dst[e]; } else { s = d = e - EE; }
}

// alpha = leaky0.2(a_src[s]+a_dst[d]+ea.w_eh)  + segment max
__global__ void k_alpha(const int* __restrict__ src, const int* __restrict__ dst,
                        const float* __restrict__ ea, const float* __restrict__ loop_attr,
                        const float* __restrict__ a_src, const float* __restrict__ a_dst,
                        const float* __restrict__ weh, float* __restrict__ alpha,
                        float* __restrict__ amax) {
    int e = blockIdx.x * blockDim.x + threadIdx.x;
    if (e >= ENTOT) return;
    int s, d; edge_sd(e, src, dst, s, d);
    const float* eav = (e < EE) ? (ea + (size_t)e * 8) : (loop_attr + (size_t)(e - EE) * 8);
    for (int hh = 0; hh < 4; ++hh) {
        float ae = 0.f;
        for (int j = 0; j < 8; ++j) ae += eav[j] * weh[j * 4 + hh];
        float a = lrelu(a_src[s * 4 + hh] + a_dst[d * 4 + hh] + ae, ASLOPE);
        alpha[(size_t)e * 4 + hh] = a;
        atomicMaxF(&amax[d * 4 + hh], a);
    }
}

// ex = exp(alpha - amax[d]) in-place + denom
__global__ void k_expden(const int* __restrict__ src, const int* __restrict__ dst,
                         float* __restrict__ alpha, const float* __restrict__ amax,
                         float* __restrict__ denom) {
    int e = blockIdx.x * blockDim.x + threadIdx.x;
    if (e >= ENTOT) return;
    int s, d; edge_sd(e, src, dst, s, d);
    for (int hh = 0; hh < 4; ++hh) {
        float v = expf(alpha[(size_t)e * 4 + hh] - amax[d * 4 + hh]);
        alpha[(size_t)e * 4 + hh] = v;
        atomicAdd(&denom[d * 4 + hh], v);
    }
}

// out[d,:] += xp[s,:] * w[e,h]   (one block of 128 threads per edge, 4 ch each)
__global__ void k_aggregate(const int* __restrict__ src, const int* __restrict__ dst,
                            const float* __restrict__ ex, const float* __restrict__ denom,
                            const float* __restrict__ xp, float* __restrict__ out) {
    int e = blockIdx.x;
    int s, d; edge_sd(e, src, dst, s, d);
    int t = threadIdx.x;                         // 0..127
    for (int hh = 0; hh < 4; ++hh) {
        float w = ex[(size_t)e * 4 + hh] / denom[d * 4 + hh];
        int c = hh * 128 + t;
        atomicAdd(&out[(size_t)d * HCC + c], xp[(size_t)s * HCC + c] * w);
    }
}

// h = leaky(out + bias) in place, optional bf16 copy
__global__ void k_finalize(float* __restrict__ out, const float* __restrict__ bias,
                           unsigned short* __restrict__ hbf) {
    size_t idx = (size_t)blockIdx.x * blockDim.x + threadIdx.x;
    if (idx >= (size_t)NNODE * HCC) return;
    float v = lrelu(out[idx] + bias[idx & (HCC - 1)], LRELU);
    out[idx] = v;
    if (hbf) hbf[idx] = f2bf(v);
}

// local map build: h2(f32 nodes) -> lm [B,66,130,512] bf16 channels-last
__global__ void k_build_lm(const float* __restrict__ h2, unsigned short* __restrict__ lm) {
    size_t idx = (size_t)blockIdx.x * blockDim.x + threadIdx.x;
    if (idx >= (size_t)BB * 66 * 130 * 512) return;
    int c = idx & 511;
    size_t r1 = idx >> 9;
    int j = (int)(r1 % 130); size_t r2 = r1 / 130;
    int r = (int)(r2 % 66);  int b = (int)(r2 / 66);
    int sy, sx;
    if (r == 0)       { sy = 1;     sx = (63 + j) & 127; }
    else if (r == 65) { sy = 62;    sx = (63 + j) & 127; }
    else              { sy = r - 1; sx = (j + 127) & 127; }
    size_t node = (size_t)b * NPB + 2 + sy * WWW + sx;
    lm[idx] = f2bf(h2[node * HCC + c]);
}

// zin[:, 0:512] = bf16(g)
__global__ void k_zin_g(const float* __restrict__ h2, unsigned short* __restrict__ zin) {
    size_t idx = (size_t)blockIdx.x * blockDim.x + threadIdx.x;
    if (idx >= (size_t)PIX * 512) return;
    int c = idx & 511;
    size_t m = idx >> 9;
    int b = (int)(m >> 13), p = (int)(m & (HWPIX - 1));
    size_t node = (size_t)b * NPB + 2 + p;
    zin[m * 640 + c] = f2bf(h2[node * HCC + c]);
}

// out[b,t,y,x] = z3[m,t] + x[b, 78+t, y, x]
__global__ void k_residual(const float* __restrict__ z3, const float* __restrict__ x,
                           float* __restrict__ outp) {
    int idx = blockIdx.x * blockDim.x + threadIdx.x;
    if (idx >= BB * TGTN * HWPIX) return;
    int xx = idx & 127, yy = (idx >> 7) & 63;
    int t = (idx / HWPIX) % TGTN, b = idx / (TGTN * HWPIX);
    size_t m = (size_t)b * HWPIX + yy * WWW + xx;
    outp[idx] = z3[m * 48 + t] + x[(((size_t)b * CXN + 78 + t) * HHH + yy) * WWW + xx];
}

// ---------------- host orchestration ----------------
static inline dim3 g1d(size_t n) { return dim3((unsigned)((n + 255) / 256)); }

extern "C" void kernel_launch(void* const* d_in, const int* in_sizes, int n_in,
                              void* d_out, int out_size, void* d_ws, size_t ws_size,
                              hipStream_t stream) {
    const float* x      = (const float*)d_in[0];
    const float* x_cons = (const float*)d_in[1];
    const int*   ei     = (const int*)d_in[2];
    const float* eattr  = (const float*)d_in[3];
    const float* g1_W  = (const float*)d_in[4];
    const float* g1_as = (const float*)d_in[5];
    const float* g1_ad = (const float*)d_in[6];
    const float* g1_We = (const float*)d_in[7];
    const float* g1_ae = (const float*)d_in[8];
    const float* g1_b  = (const float*)d_in[9];
    const float* g2_W  = (const float*)d_in[10];
    const float* g2_as = (const float*)d_in[11];
    const float* g2_ad = (const float*)d_in[12];
    const float* g2_We = (const float*)d_in[13];
    const float* g2_ae = (const float*)d_in[14];
    const float* g2_b  = (const float*)d_in[15];
    const float* conv_w = (const float*)d_in[16];
    const float* conv_b = (const float*)d_in[17];
    const float* bn_g   = (const float*)d_in[18];
    const float* bn_b   = (const float*)d_in[19];
    const float* m_w1 = (const float*)d_in[20];
    const float* m_b1 = (const float*)d_in[21];
    const float* m_w2 = (const float*)d_in[22];
    const float* m_b2 = (const float*)d_in[23];
    const float* m_w3 = (const float*)d_in[24];
    const float* m_b3 = (const float*)d_in[25];
    const int* src = ei;
    const int* dst = ei + EE;
    float* outp = (float*)d_out;

    // ---- workspace layout (with phase reuse) ----
    char* ws = (char*)d_ws;
    size_t o = 0;
    auto alloc = [&](size_t bytes) { void* p = ws + o; o = (o + bytes + 255) & ~(size_t)255; return p; };
    float*          xp     = (float*)alloc((size_t)NNODE * HCC * 4);         // both layers
    float*          outb   = (float*)alloc((size_t)NNODE * HCC * 4);         // agg / h1 / h2
    unsigned short* nodesC = (unsigned short*)alloc((size_t)BB * 66 * 130 * 512 * 2); // nodes_bf then lm
    unsigned short* hbfD   = (unsigned short*)alloc((size_t)NPADR * HCC * 2);         // h1bf then z1bf
    unsigned short* zin    = (unsigned short*)alloc((size_t)PIX * 640 * 2);
    unsigned short* z2bf   = (unsigned short*)alloc((size_t)PIX * 256 * 2);
    float*          z3     = (float*)alloc((size_t)PIX * 48 * 4);
    unsigned short* W1t    = (unsigned short*)alloc((size_t)512 * 128 * 2);
    unsigned short* W2t    = (unsigned short*)alloc((size_t)512 * 512 * 2);
    unsigned short* wct    = (unsigned short*)alloc((size_t)9 * 128 * 512 * 2);
    unsigned short* m1t    = (unsigned short*)alloc((size_t)512 * 640 * 2);
    unsigned short* m2t    = (unsigned short*)alloc((size_t)256 * 512 * 2);
    unsigned short* m3t    = (unsigned short*)alloc((size_t)48 * 256 * 2);
    float*          weh1   = (float*)alloc(128);
    float*          weh2   = (float*)alloc(128);
    float*          a_src  = (float*)alloc((size_t)NNODE * 4 * 4);
    float*          a_dst  = (float*)alloc((size_t)NNODE * 4 * 4);
    float*          amax   = (float*)alloc((size_t)NNODE * 4 * 4);
    float*          denom  = (float*)alloc((size_t)NNODE * 4 * 4);
    float*          cnt    = (float*)alloc((size_t)NNODE * 4);
    float*          lattr  = (float*)alloc((size_t)NNODE * 8 * 4);
    float*          alpha  = (float*)alloc((size_t)ENTOT * 4 * 4);
    unsigned short* lm   = nodesC;  // reuse after GEMM1 consumes node feats
    unsigned short* z1bf = hbfD;    // reuse after GEMM2 consumes h1bf

    const dim3 B256(256);

    // ---- weight prep ----
    k_wtrans<<<g1d((size_t)512 * 128), B256, 0, stream>>>(g1_W, 123, 512, W1t, 128, 512);
    k_wtrans<<<g1d((size_t)512 * 512), B256, 0, stream>>>(g2_W, 512, 512, W2t, 512, 512);
    k_wtrans<<<g1d((size_t)512 * 640), B256, 0, stream>>>(m_w1, 640, 512, m1t, 640, 512);
    k_wtrans<<<g1d((size_t)256 * 512), B256, 0, stream>>>(m_w2, 512, 256, m2t, 512, 256);
    k_wtrans<<<g1d((size_t)48 * 256),  B256, 0, stream>>>(m_w3, 256, 39,  m3t, 256, 48);
    k_convtrans<<<g1d((size_t)9 * 128 * 512), B256, 0, stream>>>(conv_w, wct);
    k_weh<<<1, 32, 0, stream>>>(g1_We, g1_ae, weh1);
    k_weh<<<1, 32, 0, stream>>>(g2_We, g2_ae, weh2);

    // ---- node features + pad-row zeroing of h1bf + self-loop attrs ----
    k_build_nodes<<<g1d((size_t)NPADR * 128), B256, 0, stream>>>(x, x_cons, nodesC);
    k_fill_u32<<<g1d((size_t)(NPADR - NNODE) * HCC / 2), B256, 0, stream>>>(
        (unsigned int*)(hbfD + (size_t)NNODE * HCC), (size_t)(NPADR - NNODE) * HCC / 2, 0u);
    k_fill_u32<<<g1d(NNODE), B256, 0, stream>>>((unsigned int*)cnt, NNODE, 0u);
    k_fill_u32<<<g1d((size_t)NNODE * 8), B256, 0, stream>>>((unsigned int*)lattr, (size_t)NNODE * 8, 0u);
    k_loopsum<<<g1d(EE), B256, 0, stream>>>(dst, eattr, cnt, lattr);
    k_loopdiv<<<g1d((size_t)NNODE * 8), B256, 0, stream>>>(lattr, cnt);

    auto gat_layer = [&](const unsigned short* Abf, int lda, int K,
                         const unsigned short* Wt, const float* as, const float* ad,
                         const float* weh, const float* bias, unsigned short* hbf) {
        dim3 gg((NNODE + 63) / 64, HCC / 16);
        k_gemm_wmma<<<gg, dim3(128), 0, stream>>>(Abf, lda, Wt, K, nullptr,
                                                  xp, HCC, nullptr, 0, NNODE, HCC, K, 0);
        k_attdot<<<g1d((size_t)NNODE * 4), B256, 0, stream>>>(xp, as, ad, a_src, a_dst);
        k_fill_u32<<<g1d((size_t)NNODE * 4), B256, 0, stream>>>((unsigned int*)amax, (size_t)NNODE * 4, 0xFF800000u);
        k_alpha<<<g1d(ENTOT), B256, 0, stream>>>(src, dst, eattr, lattr, a_src, a_dst, weh, alpha, amax);
        k_fill_u32<<<g1d((size_t)NNODE * 4), B256, 0, stream>>>((unsigned int*)denom, (size_t)NNODE * 4, 0u);
        k_expden<<<g1d(ENTOT), B256, 0, stream>>>(src, dst, alpha, amax, denom);
        k_fill_u32<<<g1d((size_t)NNODE * HCC), B256, 0, stream>>>((unsigned int*)outb, (size_t)NNODE * HCC, 0u);
        k_aggregate<<<dim3(ENTOT), dim3(128), 0, stream>>>(src, dst, alpha, denom, xp, outb);
        k_finalize<<<g1d((size_t)NNODE * HCC), B256, 0, stream>>>(outb, bias, hbf);
    };

    gat_layer(nodesC, 128, 128, W1t, g1_as, g1_ad, weh1, g1_b, hbfD);   // layer 1 -> h1 + h1bf
    gat_layer(hbfD, HCC, HCC, W2t, g2_as, g2_ad, weh2, g2_b, nullptr);  // layer 2 -> h2 (outb)

    // ---- spatial path ----
    k_build_lm<<<g1d((size_t)BB * 66 * 130 * 512), B256, 0, stream>>>(outb, lm);
    k_zin_g<<<g1d((size_t)PIX * 512), B256, 0, stream>>>(outb, zin);
    k_conv_wmma<<<dim3(PIX / 64, 128 / 16), dim3(128), 0, stream>>>(lm, wct, conv_b, bn_g, bn_b, zin);

    // ---- MLP ----
    k_gemm_wmma<<<dim3(PIX / 64, 512 / 16), dim3(128), 0, stream>>>(
        zin, 640, m1t, 640, m_b1, nullptr, 0, z1bf, 512, PIX, 512, 640, 1);
    k_gemm_wmma<<<dim3(PIX / 64, 256 / 16), dim3(128), 0, stream>>>(
        z1bf, 512, m2t, 512, m_b2, nullptr, 0, z2bf, 256, PIX, 256, 512, 1);
    k_gemm_wmma<<<dim3(PIX / 64, 48 / 16), dim3(128), 0, stream>>>(
        z2bf, 256, m3t, 256, m_b3, z3, 48, nullptr, 0, PIX, TGTN, 256, 0);

    k_residual<<<g1d((size_t)BB * TGTN * HWPIX), B256, 0, stream>>>(z3, x, outp);
}
